// ESN_17257178595342
// MI455X (gfx1250) — matrix operand order repaired and checked by reading, not compiled, not run on hardware
//
#include <hip/hip_runtime.h>
#include <hip/hip_bf16.h>

#define SEQ      512
#define BATCH    32
#define IN_DIM   128
#define HID      2048
#define OUT_DIM  128
#define NLAYER   3
#define NWG      32
#define NTHREADS 256
#define KCH      512
#define A_LEAK   0.95f

#define AS1 __attribute__((address_space(1)))
#define AS3 __attribute__((address_space(3)))

typedef __attribute__((ext_vector_type(16))) __bf16 v16bf;
typedef __attribute__((ext_vector_type(8)))  __bf16 v8bf;
typedef __attribute__((ext_vector_type(8)))  float  v8f;
typedef __attribute__((ext_vector_type(4)))  int    v4i;   // native vector for builtins

union ABFrag { v16bf v; v8bf h[2]; };
struct AccPair { v8f a0, a1; };

typedef __hip_bfloat16 bf16s;

#if __has_builtin(__builtin_amdgcn_global_load_async_to_lds_b128)
#define HAVE_ASYNC_LDS 1
#else
#define HAVE_ASYNC_LDS 0
#endif

// ---------------- workspace layout (bytes, all 256-aligned) ----------------
#define OFF_BAR   0ull
#define OFF_HF    256ull                                   // [2][3][32][2048] f32
#define OFF_HB    (OFF_HF   + 2ull*NLAYER*BATCH*HID*4)     // [2][3][32][2048] bf16
#define OFF_XB    (OFF_HB   + 2ull*NLAYER*BATCH*HID*2)     // [512][32][128]  bf16
#define OFF_WIH0  (OFF_XB   + (unsigned long long)SEQ*BATCH*IN_DIM*2)
#define OFF_WHH0  (OFF_WIH0 + (unsigned long long)HID*IN_DIM*2)
#define OFF_WIH1  (OFF_WHH0 + (unsigned long long)HID*HID*2)
#define OFF_WHH1  (OFF_WIH1 + (unsigned long long)HID*HID*2)
#define OFF_WIH2  (OFF_WHH1 + (unsigned long long)HID*HID*2)
#define OFF_WHH2  (OFF_WIH2 + (unsigned long long)HID*HID*2)
#define OFF_WOUT  (OFF_WHH2 + (unsigned long long)HID*HID*2)
#define WS_TOTAL  (OFF_WOUT + (unsigned long long)OUT_DIM*(NLAYER*HID)*2)

// ---------------- prep kernels ----------------

__global__ void esn_cvt_bf16(const float* __restrict__ s, bf16s* __restrict__ d, int n) {
  for (int i = blockIdx.x * blockDim.x + threadIdx.x; i < n; i += gridDim.x * blockDim.x)
    d[i] = __float2bfloat16(s[i]);
}

// Pack W[N][K] (row-major fp32) into WMMA-B fragments:
// elem index = ((nt*(K/32)+kc)*32 + lane)*16 ; lane: n = nt*16 + (lane&15),
// k = kc*32 + (lane>>4)*16 + j  (j = 0..15, contiguous along K).
__global__ void esn_pack_b(const float* __restrict__ src, bf16s* __restrict__ dst,
                           int N, int K) {
  int idx = blockIdx.x * blockDim.x + threadIdx.x;
  int total = (N * K) >> 4;
  if (idx >= total) return;
  int lane = idx & 31;
  int kcn  = idx >> 5;            // nt*(K/32) + kc
  int kdiv = K >> 5;
  int kc   = kcn % kdiv;
  int nt   = kcn / kdiv;
  int n     = nt * 16 + (lane & 15);
  int kbase = kc * 32 + (lane >> 4) * 16;
  const float* s = src + (size_t)n * K + kbase;
  bf16s* d = dst + ((size_t)idx << 4);
#pragma unroll
  for (int j = 0; j < 16; ++j) d[j] = __float2bfloat16(s[j]);
}

// ---------------- device helpers ----------------

__device__ __forceinline__ void gbar(unsigned* bar, unsigned target) {
  __threadfence();
  __syncthreads();
  if (threadIdx.x == 0) {
    atomicAdd(bar, 1u);
    while (__hip_atomic_load(bar, __ATOMIC_RELAXED, __HIP_MEMORY_SCOPE_AGENT) < target)
      __builtin_amdgcn_s_sleep(2);
  }
  __syncthreads();
  __threadfence();
}

template<int N>
__device__ __forceinline__ void wait_async_n() {
#if HAVE_ASYNC_LDS
#if __has_builtin(__builtin_amdgcn_s_wait_asynccnt)
  __builtin_amdgcn_s_wait_asynccnt(N);
#else
  asm volatile("s_wait_asynccnt %0" :: "i"(N) : "memory");
#endif
#endif
}

// Issue async gather of 32 rows x KC bf16 cols (row stride RS elems) into LDS [32][KC].
// No wait: completion is claimed later via wait_async_n<> (in-order per wave).
template<int RS, int KC>
__device__ __forceinline__ void stage_issue(const bf16s* __restrict__ g, int cs,
                                            bf16s* __restrict__ s) {
  constexpr int VPR  = KC >> 3;            // 16B vectors per row
  constexpr int VECS = BATCH * VPR;
  constexpr int ITER = VECS / NTHREADS;
#if HAVE_ASYNC_LDS
  AS3 unsigned char* s3 = (AS3 unsigned char*)s;
#pragma unroll
  for (int k = 0; k < ITER; ++k) {
    const int i = k * NTHREADS + threadIdx.x;
    const int r = i / VPR, c = i % VPR;
    __builtin_amdgcn_global_load_async_to_lds_b128(
        (AS1 v4i*)(g + (size_t)r * RS + cs + (c << 3)),
        (AS3 v4i*)(s3 + (size_t)i * 16), 0, 0);
  }
#else
  v4i tmp[ITER];
#pragma unroll
  for (int k = 0; k < ITER; ++k) {
    const int i = k * NTHREADS + threadIdx.x;
    const int r = i / VPR, c = i % VPR;
    tmp[k] = *(const v4i*)(g + (size_t)r * RS + cs + (c << 3));
  }
#pragma unroll
  for (int k = 0; k < ITER; ++k)
    ((v4i*)s)[k * NTHREADS + threadIdx.x] = tmp[k];
#endif
}

// Dual-accumulator K-chunk GEMM: A from LDS [32][KC], B from packed weights.
// Alternating accumulators break the WMMA->WMMA RAW chain (no v_nop stalls).
template<int KC>
__device__ __forceinline__ void accum_t(AccPair& acc, const bf16s* __restrict__ sA,
                                        const bf16s* __restrict__ wrow,
                                        int arow, int khalf) {
  const bf16s* ap0 = sA + arow * KC + khalf * 16;
#pragma unroll 2
  for (int c = 0; c < KC; c += 64) {
    ABFrag a0, b0, a1, b1;
    a0.h[0] = *(const v8bf*)(ap0 + c);
    a0.h[1] = *(const v8bf*)(ap0 + c + 8);
    b0.v    = *(const v16bf*)(wrow + (size_t)(c >> 5) * 512);
    a1.h[0] = *(const v8bf*)(ap0 + c + 32);
    a1.h[1] = *(const v8bf*)(ap0 + c + 40);
    b1.v    = *(const v16bf*)(wrow + (size_t)(c >> 5) * 512 + 512);
    __builtin_prefetch(wrow + (size_t)(c >> 5) * 512 + (size_t)KCH * 16, 0, 1);
    acc.a0 = __builtin_amdgcn_wmma_f32_16x16x32_bf16(false, a0.v, false, b0.v,
                                                     (short)0, acc.a0, false, false);
    acc.a1 = __builtin_amdgcn_wmma_f32_16x16x32_bf16(false, a1.v, false, b1.v,
                                                     (short)0, acc.a1, false, false);
  }
}

// One layer's full GEMM (input + recurrent terms) as a single double-buffered
// chunk pipeline. Chunk 0 must already be in flight to sA0 when called
// (issued before the preceding grid barrier so the load hides under it).
// RECFIRST: recurrent term first (its source is stable across the barrier).
template<int KIN, bool RECFIRST>
__device__ __forceinline__ AccPair layer_gemm(
    const bf16s* __restrict__ gin,  const bf16s* __restrict__ wih,
    const bf16s* __restrict__ grec, const bf16s* __restrict__ whh,
    bf16s* __restrict__ sA0, bf16s* __restrict__ sA1,
    int ntile, int arow, int khalf, int lane) {
  constexpr int KC1  = (KIN < KCH) ? KIN : KCH;   // input-chunk width
  constexpr int NCH1 = KIN / KC1;                 // input chunks
  constexpr int NCH2 = HID / KCH;                 // recurrent chunks
  constexpr int NTOT = NCH1 + NCH2;
  AccPair acc;
  acc.a0 = v8f{0.f, 0.f, 0.f, 0.f, 0.f, 0.f, 0.f, 0.f};
  acc.a1 = v8f{0.f, 0.f, 0.f, 0.f, 0.f, 0.f, 0.f, 0.f};
#pragma unroll
  for (int j = 0; j < NTOT; ++j) {
    bf16s* curb = (j & 1) ? sA1 : sA0;
    bf16s* nxtb = (j & 1) ? sA0 : sA1;
    if (j + 1 < NTOT) {                 // issue chunk j+1 (always 512-wide: 8 ops)
      const int jj = j + 1;
      if (RECFIRST ? (jj < NCH2) : (jj >= NCH1)) {
        const int ri = RECFIRST ? jj : (jj - NCH1);
        stage_issue<HID, KCH>(grec, ri * KCH, nxtb);
      } else {
        const int ci = RECFIRST ? (jj - NCH2) : jj;
        stage_issue<KIN, KC1>(gin, ci * KC1, nxtb);
      }
      wait_async_n<8>();                // in-order => chunk j has fully landed
    } else {
      wait_async_n<0>();
    }
    __syncthreads();
    if (RECFIRST ? (j < NCH2) : (j >= NCH1)) {
      const int ri = RECFIRST ? j : (j - NCH1);
      const bf16s* wrow = whh +
          (((size_t)ntile * (HID >> 5) + (size_t)ri * (KCH >> 5)) << 9) + lane * 16;
      accum_t<KCH>(acc, curb, wrow, arow, khalf);
    } else {
      const int ci = RECFIRST ? (j - NCH2) : j;
      const bf16s* wrow = wih +
          (((size_t)ntile * (KIN >> 5) + (size_t)ci * (KC1 >> 5)) << 9) + lane * 16;
      accum_t<KC1>(acc, curb, wrow, arow, khalf);
    }
    __syncthreads();
  }
  return acc;
}

// bias + tanh + leaky blend; write fp32 master + bf16 mirror
__device__ __forceinline__ void epilogue(const AccPair& acc, const float* __restrict__ bias,
                                         const float* __restrict__ hOld,
                                         float* __restrict__ hNew, bf16s* __restrict__ hNewB,
                                         int ntile, int l16, int mbase, int khalf) {
  const int   n  = ntile * 16 + l16;
  const float bs = bias[n];
#pragma unroll
  for (int i = 0; i < 8; ++i) {
    const int b   = mbase + khalf * 8 + i;       // C-frag: VGPR i, lane half
    const int idx = b * HID + n;
    const float hv = (1.f - A_LEAK) * hOld[idx] +
                     A_LEAK * tanhf(acc.a0[i] + acc.a1[i] + bs);
    hNew[idx]  = hv;
    hNewB[idx] = __float2bfloat16(hv);
  }
}

// ---------------- persistent ESN kernel ----------------

__global__ __launch_bounds__(NTHREADS) void esn_persistent(
    const bf16s* __restrict__ xb,       // [SEQ][B][IN] bf16
    float*       __restrict__ hf,       // [2][3][B][H] f32 (state master)
    bf16s*       __restrict__ hb,       // [2][3][B][H] bf16 (matmul mirror)
    const bf16s* __restrict__ wih0, const bf16s* __restrict__ whh0,
    const bf16s* __restrict__ wih1, const bf16s* __restrict__ whh1,
    const bf16s* __restrict__ wih2, const bf16s* __restrict__ whh2,
    const bf16s* __restrict__ wout,
    const float* __restrict__ b0, const float* __restrict__ b1,
    const float* __restrict__ b2, const float* __restrict__ bo,
    const int*   __restrict__ washout_p,
    float*       __restrict__ out,      // [SEQ-wo][B][OUT] then [3][B][H]
    unsigned*    __restrict__ bar) {
  extern __shared__ unsigned char smem[];
  bf16s* sA0  = (bf16s*)smem;                                  // [32][KCH] (32 KB)
  bf16s* sA1  = (bf16s*)(smem + (size_t)BATCH * KCH * 2);      // [32][KCH] (32 KB)
  float* sred = (float*)(smem + (size_t)2 * BATCH * KCH * 2);  // [8][256]  (8 KB)

  const int wg    = blockIdx.x;
  const int tid   = threadIdx.x;
  const int wave  = tid >> 5;
  const int lane  = tid & 31;
  const int l16   = lane & 15;
  const int khalf = lane >> 4;
  const int mbase = (wave >> 2) * 16;        // batch tile: 0 or 16
  const int ntile = wg * 4 + (wave & 3);     // 0..127
  const int arow  = mbase + l16;             // A row (batch index) for this lane
  const int wo    = *washout_p;

  constexpr size_t SZ = (size_t)BATCH * HID;
  unsigned phase = 0;

  // pre-issue layer-0 input chunk for t=0 (x is always available)
  stage_issue<IN_DIM, IN_DIM>(xb, 0, sA0);

  for (int t = 0; t < SEQ; ++t) {
    const int cur = t & 1, nxt = cur ^ 1;
    const size_t c0 = (size_t)(cur * NLAYER) * SZ, n0 = (size_t)(nxt * NLAYER) * SZ;

    // ---- layer 0 (input term first: x-chunk already in flight) ----
    {
      AccPair a = layer_gemm<IN_DIM, false>(xb + (size_t)t * BATCH * IN_DIM, wih0,
                                            hb + c0 + 0 * SZ, whh0,
                                            sA0, sA1, ntile, arow, khalf, lane);
      epilogue(a, b0, hf + c0 + 0 * SZ, hf + n0 + 0 * SZ, hb + n0 + 0 * SZ,
               ntile, l16, mbase, khalf);
    }
    stage_issue<HID, KCH>(hb + c0 + 1 * SZ, 0, sA0);   // layer-1 rec chunk 0
    gbar(bar, (++phase) * NWG);

    // ---- layer 1 (recurrent term first: its chunk 0 rode the barrier) ----
    {
      AccPair a = layer_gemm<HID, true>(hb + n0 + 0 * SZ, wih1,
                                        hb + c0 + 1 * SZ, whh1,
                                        sA0, sA1, ntile, arow, khalf, lane);
      epilogue(a, b1, hf + c0 + 1 * SZ, hf + n0 + 1 * SZ, hb + n0 + 1 * SZ,
               ntile, l16, mbase, khalf);
    }
    stage_issue<HID, KCH>(hb + c0 + 2 * SZ, 0, sA0);   // layer-2 rec chunk 0
    gbar(bar, (++phase) * NWG);

    // ---- layer 2 ----
    {
      AccPair a = layer_gemm<HID, true>(hb + n0 + 1 * SZ, wih2,
                                        hb + c0 + 2 * SZ, whh2,
                                        sA0, sA1, ntile, arow, khalf, lane);
      epilogue(a, b2, hf + c0 + 2 * SZ, hf + n0 + 2 * SZ, hb + n0 + 2 * SZ,
               ntile, l16, mbase, khalf);
    }
    if (t + 1 < SEQ)                                    // next step's x chunk
      stage_issue<IN_DIM, IN_DIM>(xb + (size_t)(t + 1) * BATCH * IN_DIM, 0, sA0);
    gbar(bar, (++phase) * NWG);

    // ---- fused output projection: out[t] = [h0,h1,h2] @ Wout^T + b_out ----
    if (t >= wo && wg < 16) {
      const int mt = wg & 1, nt2 = wg >> 1;
      const int mb2 = mt * 16, nb2 = nt2 * 16;
      const int row2 = mb2 + l16;
      AccPair oa;
      oa.a0 = v8f{0.f, 0.f, 0.f, 0.f, 0.f, 0.f, 0.f, 0.f};
      oa.a1 = v8f{0.f, 0.f, 0.f, 0.f, 0.f, 0.f, 0.f, 0.f};
#pragma unroll 4
      for (int c2 = 0; c2 < 24; c2 += 2) {       // 8 waves x 768-wide K slices = 6144
#pragma unroll
        for (int u = 0; u < 2; ++u) {
          const int kk   = wave * 768 + (c2 + u) * 32;
          const int sel  = kk >> 11;             // which layer's h
          const int koff = kk & (HID - 1);
          const bf16s* ap = hb + n0 + (size_t)sel * SZ +
                            (size_t)row2 * HID + koff + khalf * 16;
          ABFrag a, b;
          a.h[0] = *(const v8bf*)ap;
          a.h[1] = *(const v8bf*)(ap + 8);
          b.v = *(const v16bf*)(wout + (((size_t)nt2 * 192 + (kk >> 5)) << 9) + lane * 16);
          v8f& dst = u ? oa.a1 : oa.a0;
          dst = __builtin_amdgcn_wmma_f32_16x16x32_bf16(false, a.v, false, b.v,
                                                        (short)0, dst, false, false);
        }
      }
      __syncthreads();
#pragma unroll
      for (int i = 0; i < 8; ++i)
        sred[wave * 256 + lane * 8 + i] = oa.a0[i] + oa.a1[i];
      __syncthreads();
      {
        const int l2 = tid >> 3, i2 = tid & 7;
        float v = 0.f;
#pragma unroll
        for (int w = 0; w < 8; ++w) v += sred[w * 256 + l2 * 8 + i2];
        const int kh2 = l2 >> 4;
        const int b   = mb2 + kh2 * 8 + i2;
        const int n   = nb2 + (l2 & 15);
        out[((size_t)(t - wo) * BATCH + b) * OUT_DIM + n] = v + bo[n];
      }
    }
  }

  // final hidden: state after step SEQ-1 lives in buffer (SEQ & 1) = 0
  float* hid = out + (size_t)(SEQ - wo) * BATCH * OUT_DIM;
  for (int i = wg * NTHREADS + tid; i < NLAYER * BATCH * HID; i += NWG * NTHREADS)
    hid[i] = hf[i];
}

// ---------------- host ----------------

extern "C" void kernel_launch(void* const* d_in, const int* in_sizes, int n_in,
                              void* d_out, int out_size, void* d_ws, size_t ws_size,
                              hipStream_t stream) {
  (void)in_sizes; (void)n_in; (void)out_size;
  if (ws_size < WS_TOTAL) return;

  const float* x     = (const float*)d_in[0];
  const float* w_ih0 = (const float*)d_in[1];
  const float* b_ih0 = (const float*)d_in[2];
  const float* w_hh0 = (const float*)d_in[3];
  const float* w_ih1 = (const float*)d_in[4];
  const float* b_ih1 = (const float*)d_in[5];
  const float* w_hh1 = (const float*)d_in[6];
  const float* w_ih2 = (const float*)d_in[7];
  const float* b_ih2 = (const float*)d_in[8];
  const float* w_hh2 = (const float*)d_in[9];
  const float* w_out = (const float*)d_in[10];
  const float* b_out = (const float*)d_in[11];
  const int*   wo    = (const int*)d_in[12];

  char* ws = (char*)d_ws;
  unsigned* bar = (unsigned*)(ws + OFF_BAR);
  float*    hf  = (float*)(ws + OFF_HF);
  bf16s*    hb  = (bf16s*)(ws + OFF_HB);
  bf16s*    xb  = (bf16s*)(ws + OFF_XB);
  bf16s*    p_ih0 = (bf16s*)(ws + OFF_WIH0);
  bf16s*    p_hh0 = (bf16s*)(ws + OFF_WHH0);
  bf16s*    p_ih1 = (bf16s*)(ws + OFF_WIH1);
  bf16s*    p_hh1 = (bf16s*)(ws + OFF_WHH1);
  bf16s*    p_ih2 = (bf16s*)(ws + OFF_WIH2);
  bf16s*    p_hh2 = (bf16s*)(ws + OFF_WHH2);
  bf16s*    p_out = (bf16s*)(ws + OFF_WOUT);

  // zero barrier counter + both h buffers (fp32 + bf16) — contiguous region
  (void)hipMemsetAsync(ws, 0, (size_t)OFF_XB, stream);

  // input -> bf16
  esn_cvt_bf16<<<1024, NTHREADS, 0, stream>>>(x, xb, SEQ * BATCH * IN_DIM);

  // pack weights into WMMA-B fragment order (bf16)
  auto packs = [&](const float* s, bf16s* d, int N, int K) {
    int thr = (N * K) >> 4;
    esn_pack_b<<<(thr + NTHREADS - 1) / NTHREADS, NTHREADS, 0, stream>>>(s, d, N, K);
  };
  packs(w_ih0, p_ih0, HID, IN_DIM);
  packs(w_hh0, p_hh0, HID, HID);
  packs(w_ih1, p_ih1, HID, HID);
  packs(w_hh1, p_hh1, HID, HID);
  packs(w_ih2, p_ih2, HID, HID);
  packs(w_hh2, p_hh2, HID, HID);
  packs(w_out, p_out, OUT_DIM, NLAYER * HID);

  const size_t smem_bytes = (size_t)2 * BATCH * KCH * 2 + 8 * 256 * 4;  // 72 KB
  esn_persistent<<<NWG, NTHREADS, smem_bytes, stream>>>(
      xb, hf, hb, p_ih0, p_hh0, p_ih1, p_hh1, p_ih2, p_hh2, p_out,
      b_ih0, b_ih1, b_ih2, b_out, wo, (float*)d_out, bar);
}
